// ModularRegimeRNN_11785390260814
// MI455X (gfx1250) — compile-verified, matrix operand order checked by reading, not run on hardware
//
#include <hip/hip_runtime.h>
#include <stdint.h>

typedef int64_t i64;
typedef __bf16 bf16;
typedef __attribute__((ext_vector_type(16))) __bf16 bf16x16;
typedef __attribute__((ext_vector_type(8)))  __bf16 bf16x8;
typedef __attribute__((ext_vector_type(8)))  float  f32x8;

#define cB  4096
#define cD  512
#define cH  1024
#define cK  8
#define cO  256
#define cNH 4
#define cHD 256
#define c3H 3072

// ---------- small numeric helpers ----------
__device__ __forceinline__ bf16 f2bf(float f) {
  uint32_t u = __builtin_bit_cast(uint32_t, f);
  u += 0x7FFFu + ((u >> 16) & 1u);          // round-to-nearest-even
  uint16_t h = (uint16_t)(u >> 16);
  return __builtin_bit_cast(bf16, h);
}
__device__ __forceinline__ float bf2f(bf16 b) {
  uint16_t s = __builtin_bit_cast(uint16_t, b);
  uint32_t u = ((uint32_t)s) << 16;
  return __builtin_bit_cast(float, u);
}
__device__ __forceinline__ float sigmoidf_(float x) { return 1.0f / (1.0f + expf(-x)); }

// ---------- WMMA fragment helpers (CDNA5 16x16x32 bf16 layout, ISA 7.12.2) ----------
// A (16x32, 16-bit): lane l -> row = l%16, kbase = (l/16)*8;
// v16 elements 0..7 -> K = kbase+0..7, elements 8..15 -> K = kbase+16..23.
// B (32x16) loads identically when sourced from row-major W[N][K] (we compute A@W^T).
__device__ __forceinline__ bf16x16 load_frag(const bf16* __restrict__ base, i64 ld, int row0, int k0) {
  const int lane = threadIdx.x & 31;
  const int r  = row0 + (lane & 15);
  const int kb = k0 + ((lane >> 4) << 3);
  const bf16* p = base + (i64)r * ld + kb;
  bf16x8 lo = *(const bf16x8*)(p);        // K = kb   .. kb+7   (16B)
  bf16x8 hi = *(const bf16x8*)(p + 16);   // K = kb+16.. kb+23  (16B)
  bf16x16 f;
#pragma unroll
  for (int i = 0; i < 8; ++i) { f[i] = lo[i]; f[i + 8] = hi[i]; }
  return f;
}

__device__ __forceinline__ f32x8 wmma_bf16(bf16x16 a, bf16x16 b, f32x8 c) {
  // (neg_a, A, neg_b, B, c_mod, C, reuse_a, reuse_b)
  return __builtin_amdgcn_wmma_f32_16x16x32_bf16(false, a, false, b, (short)0, c, false, false);
}

// ---------- kernel 0: f32 -> bf16 cast ----------
__global__ __launch_bounds__(256)
void cast_f32_bf16_kernel(const float* __restrict__ in, bf16* __restrict__ out, i64 n) {
  i64 i = (i64)blockIdx.x * 256 + threadIdx.x;
  if (i < n) out[i] = f2bf(in[i]);
}

// ---------- kernel 1: fused GRU gates via WMMA ----------
// One wave per 32(b) x 16(h) tile of one expert. Accumulators per m-tile:
// accR = i_r + h_r, accZ = i_z + h_z, accNi = i_n, accNh = h_n (kept separate for r*h_n).
// Per K-step: 2 A fragments + 3 gate-weight fragments feed 6 WMMAs.
__global__ __launch_bounds__(32)
void gru_gates_wmma_kernel(const bf16* __restrict__ xbf, const bf16* __restrict__ hbf,
                           const bf16* __restrict__ wih, const bf16* __restrict__ whh,
                           const float* __restrict__ b_ih, const float* __restrict__ b_hh,
                           const float* __restrict__ h_prev_f, float* __restrict__ h_out) {
  const int b0 = blockIdx.x << 5;       // 32 batch rows
  const int h0 = blockIdx.y << 4;       // 16 hidden cols
  const int k  = blockIdx.z;
  const f32x8 zero = {0.f,0.f,0.f,0.f,0.f,0.f,0.f,0.f};
  f32x8 accR[2]  = {zero, zero};
  f32x8 accZ[2]  = {zero, zero};
  f32x8 accNi[2] = {zero, zero};
  f32x8 accNh[2] = {zero, zero};

  const bf16* wi = wih + (i64)k * c3H * cD;
#pragma unroll 2
  for (int d = 0; d < cD; d += 32) {
    bf16x16 a0 = load_frag(xbf, cD, b0,      d);
    bf16x16 a1 = load_frag(xbf, cD, b0 + 16, d);
    bf16x16 br = load_frag(wi,                    cD, h0, d);
    bf16x16 bz = load_frag(wi + (i64)cH * cD,     cD, h0, d);
    bf16x16 bn = load_frag(wi + (i64)2 * cH * cD, cD, h0, d);
    accR[0]  = wmma_bf16(a0, br, accR[0]);   accR[1]  = wmma_bf16(a1, br, accR[1]);
    accZ[0]  = wmma_bf16(a0, bz, accZ[0]);   accZ[1]  = wmma_bf16(a1, bz, accZ[1]);
    accNi[0] = wmma_bf16(a0, bn, accNi[0]);  accNi[1] = wmma_bf16(a1, bn, accNi[1]);
  }
  const bf16* wh = whh + (i64)k * c3H * cH;
  const bf16* hb = hbf + (i64)k * cH;            // (B,K,H): row stride K*H
#pragma unroll 2
  for (int d = 0; d < cH; d += 32) {
    bf16x16 a0 = load_frag(hb, (i64)cK * cH, b0,      d);
    bf16x16 a1 = load_frag(hb, (i64)cK * cH, b0 + 16, d);
    bf16x16 br = load_frag(wh,                    cH, h0, d);
    bf16x16 bz = load_frag(wh + (i64)cH * cH,     cH, h0, d);
    bf16x16 bn = load_frag(wh + (i64)2 * cH * cH, cH, h0, d);
    accR[0]  = wmma_bf16(a0, br, accR[0]);   accR[1]  = wmma_bf16(a1, br, accR[1]);
    accZ[0]  = wmma_bf16(a0, bz, accZ[0]);   accZ[1]  = wmma_bf16(a1, bz, accZ[1]);
    accNh[0] = wmma_bf16(a0, bn, accNh[0]);  accNh[1] = wmma_bf16(a1, bn, accNh[1]);
  }
  const int lane = threadIdx.x & 31;
  const int e  = h0 + (lane & 15);
  const int mbase = (lane >> 4) << 3;
  const float bir  = b_ih[(i64)k*c3H + e],          bhr = b_hh[(i64)k*c3H + e];
  const float biz  = b_ih[(i64)k*c3H + cH + e],     bhz = b_hh[(i64)k*c3H + cH + e];
  const float bin_ = b_ih[(i64)k*c3H + 2*cH + e],   bhn = b_hh[(i64)k*c3H + 2*cH + e];
#pragma unroll
  for (int mi = 0; mi < 2; ++mi) {
#pragma unroll
    for (int i = 0; i < 8; ++i) {
      const int brow = b0 + 16*mi + mbase + i;
      const i64 idx = ((i64)brow * cK + k) * cH + e;
      float r  = sigmoidf_(accR[mi][i] + bir + bhr);
      float z  = sigmoidf_(accZ[mi][i] + biz + bhz);
      float nn = tanhf(accNi[mi][i] + bin_ + r * (accNh[mi][i] + bhn));
      h_out[idx] = (1.0f - z) * nn + z * h_prev_f[idx];   // pre-LN; LN kernel fixes in place
    }
  }
}

// ---------- kernel 2: per-expert LayerNorm (in place) + bf16 copy ----------
__global__ __launch_bounds__(256)
void expert_ln_kernel(float* __restrict__ hio, const float* __restrict__ g,
                      const float* __restrict__ bvec, bf16* __restrict__ hbf) {
  const i64 row = blockIdx.x;                 // b*K + k
  const int k = (int)(row & (cK - 1));
  float* p = hio + row * cH;
  const int t = threadIdx.x, wid = t >> 5, lane = t & 31;
  __shared__ float rs[8], rs2[8], stats[2];
  float v[4], s = 0.f, s2 = 0.f;
#pragma unroll
  for (int j = 0; j < 4; ++j) { v[j] = p[t + j*256]; s += v[j]; s2 += v[j]*v[j]; }
#pragma unroll
  for (int off = 16; off; off >>= 1) { s += __shfl_down(s, off); s2 += __shfl_down(s2, off); }
  if (lane == 0) { rs[wid] = s; rs2[wid] = s2; }
  __syncthreads();
  if (t == 0) {
    float ts = 0.f, ts2 = 0.f;
#pragma unroll
    for (int i = 0; i < 8; ++i) { ts += rs[i]; ts2 += rs2[i]; }
    float mu  = ts * (1.0f / cH);
    float var = ts2 * (1.0f / cH) - mu * mu;
    stats[0] = mu; stats[1] = rsqrtf(var + 1e-5f);
  }
  __syncthreads();
  const float mu = stats[0], rstd = stats[1];
#pragma unroll
  for (int j = 0; j < 4; ++j) {
    const int hh = t + j*256;
    float o = (v[j] - mu) * rstd * g[(i64)k*cH + hh] + bvec[(i64)k*cH + hh];
    p[hh] = o;
    hbf[row*cH + hh] = f2bf(o);
  }
}

// ---------- kernel 3: generic WMMA GEMM  out = A(MxKd) @ W(NxKd)^T + bias ----------
// 32(M) x 64(N) register tile per wave: per K-step 2 A-frags + 4 W-frags feed 8 WMMAs.
__global__ __launch_bounds__(32)
void gemm_wmma_kernel(const bf16* __restrict__ A, i64 lda,
                      const bf16* __restrict__ W, i64 ldw,
                      const float* __restrict__ bias,
                      float* __restrict__ outF, bf16* __restrict__ outB,
                      i64 ldo, int Kd) {
  const int m0 = blockIdx.x << 5;    // 32 rows
  const int n0 = blockIdx.y << 6;    // 64 cols
  const f32x8 zero = {0.f,0.f,0.f,0.f,0.f,0.f,0.f,0.f};
  f32x8 acc[2][4];
#pragma unroll
  for (int mi = 0; mi < 2; ++mi)
#pragma unroll
    for (int j = 0; j < 4; ++j) acc[mi][j] = zero;

#pragma unroll 2
  for (int d = 0; d < Kd; d += 32) {
    if (d + 32 < Kd) {   // global_prefetch_b8 for the next K-step (mostly L2-resident)
      __builtin_prefetch(A + (i64)(m0 + (threadIdx.x & 15)) * lda + d + 32, 0, 1);
      __builtin_prefetch(W + (i64)(n0 + (threadIdx.x & 15)) * ldw + d + 32, 0, 1);
    }
    bf16x16 a0 = load_frag(A, lda, m0,      d);
    bf16x16 a1 = load_frag(A, lda, m0 + 16, d);
#pragma unroll
    for (int j = 0; j < 4; ++j) {
      bf16x16 w = load_frag(W, ldw, n0 + 16*j, d);
      acc[0][j] = wmma_bf16(a0, w, acc[0][j]);
      acc[1][j] = wmma_bf16(a1, w, acc[1][j]);
    }
  }
  const int lane = threadIdx.x & 31;
  const int nloc = lane & 15;
  const int mbase = (lane >> 4) << 3;
#pragma unroll
  for (int j = 0; j < 4; ++j) {
    const int n = n0 + 16*j + nloc;
    const float bv = bias ? bias[n] : 0.f;
#pragma unroll
    for (int mi = 0; mi < 2; ++mi) {
#pragma unroll
      for (int i = 0; i < 8; ++i) {
        const float v = acc[mi][j][i] + bv;
        const i64 idx = (i64)(m0 + 16*mi + mbase + i) * ldo + n;
        if (outF) outF[idx] = v;
        if (outB) outB[idx] = f2bf(v);
      }
    }
  }
}

// ---------- kernel 4: attention over K=8 experts, NH=4 heads ----------
__global__ __launch_bounds__(256)
void attn_kernel(const float* __restrict__ q, const bf16* __restrict__ kp,
                 const bf16* __restrict__ vp, bf16* __restrict__ ctx) {
  const i64 b = blockIdx.x;
  const int t = threadIdx.x, lane = t & 31;
  __shared__ float sc[cNH * cK];
  __shared__ float at[cNH * cK];
  if (t < cNH * cK) sc[t] = 0.f;
  __syncthreads();
  float qv[4];
#pragma unroll
  for (int j = 0; j < 4; ++j) qv[j] = q[b * cH + t + j*256];   // head(h=t+j*256) == j
  for (int kk = 0; kk < cK; ++kk) {
    const bf16* kr = kp + (b * cK + kk) * cH;
#pragma unroll
    for (int j = 0; j < 4; ++j) {
      float p = qv[j] * bf2f(kr[t + j*256]);
#pragma unroll
      for (int off = 16; off; off >>= 1) p += __shfl_down(p, off);
      if (lane == 0) atomicAdd(&sc[j * cK + kk], p);           // ds_add_f32
    }
  }
  __syncthreads();
  if (t < cNH) {
    const float scale = 0.0625f;                                // 1/sqrt(HD=256)
    float mx = -1e30f;
    for (int kk = 0; kk < cK; ++kk) { float s = sc[t*cK+kk] * scale; sc[t*cK+kk] = s; mx = fmaxf(mx, s); }
    float se = 0.f;
    for (int kk = 0; kk < cK; ++kk) { float e = expf(sc[t*cK+kk] - mx); at[t*cK+kk] = e; se += e; }
    const float inv = 1.0f / se;
    for (int kk = 0; kk < cK; ++kk) at[t*cK+kk] *= inv;
  }
  __syncthreads();
#pragma unroll
  for (int j = 0; j < 4; ++j) {
    const int hh = t + j*256;
    float c = 0.f;
    for (int kk = 0; kk < cK; ++kk) c += at[j*cK+kk] * bf2f(vp[(b*cK+kk)*cH + hh]);
    ctx[b * cH + hh] = f2bf(c);
  }
}

// ---------- kernel 5: residual + LN, gate logits, softmax, expert mix ----------
__global__ __launch_bounds__(256)
void gate_mix_kernel(const float* __restrict__ query, const float* __restrict__ attn_out,
                     const float* __restrict__ aln_g, const float* __restrict__ aln_b,
                     const float* __restrict__ W_gate, const float* __restrict__ b_gate,
                     const float* __restrict__ h_new,
                     float* __restrict__ g_t, float* __restrict__ logits,
                     bf16* __restrict__ mixed_bf) {
  const i64 b = blockIdx.x;
  const int t = threadIdx.x, wid = t >> 5, lane = t & 31;
  __shared__ float rs[8], rs2[8], stats[2], lg[cK], gts[cK];
  float v[4], s = 0.f, s2 = 0.f;
#pragma unroll
  for (int j = 0; j < 4; ++j) {
    const int hh = t + j*256;
    v[j] = query[b*cH + hh] + attn_out[b*cH + hh];
    s += v[j]; s2 += v[j]*v[j];
  }
#pragma unroll
  for (int off = 16; off; off >>= 1) { s += __shfl_down(s, off); s2 += __shfl_down(s2, off); }
  if (lane == 0) { rs[wid] = s; rs2[wid] = s2; }
  if (t < cK) lg[t] = b_gate[t];
  __syncthreads();
  if (t == 0) {
    float ts = 0.f, ts2 = 0.f;
#pragma unroll
    for (int i = 0; i < 8; ++i) { ts += rs[i]; ts2 += rs2[i]; }
    float mu  = ts * (1.0f / cH);
    float var = ts2 * (1.0f / cH) - mu * mu;
    stats[0] = mu; stats[1] = rsqrtf(var + 1e-5f);
  }
  __syncthreads();
  const float mu = stats[0], rstd = stats[1];
  float c[4];
#pragma unroll
  for (int j = 0; j < 4; ++j) {
    const int hh = t + j*256;
    c[j] = (v[j] - mu) * rstd * aln_g[hh] + aln_b[hh];
  }
  for (int kk = 0; kk < cK; ++kk) {
    float p = 0.f;
#pragma unroll
    for (int j = 0; j < 4; ++j) p += c[j] * W_gate[(i64)kk*cH + t + j*256];
#pragma unroll
    for (int off = 16; off; off >>= 1) p += __shfl_down(p, off);
    if (lane == 0) atomicAdd(&lg[kk], p);
  }
  __syncthreads();
  if (t == 0) {
    float mx = lg[0];
    for (int kk = 1; kk < cK; ++kk) mx = fmaxf(mx, lg[kk]);
    float se = 0.f;
    for (int kk = 0; kk < cK; ++kk) { float e = expf(lg[kk] - mx); gts[kk] = e; se += e; }
    const float inv = 1.0f / se;
    for (int kk = 0; kk < cK; ++kk) {
      gts[kk] *= inv;
      g_t[b*cK + kk]    = gts[kk];
      logits[b*cK + kk] = lg[kk];
    }
  }
  __syncthreads();
#pragma unroll
  for (int j = 0; j < 4; ++j) {
    const int hh = t + j*256;
    float m = 0.f;
    for (int kk = 0; kk < cK; ++kk) m += gts[kk] * h_new[(b*cK + kk)*cH + hh];
    mixed_bf[b*cH + hh] = f2bf(m);
  }
}

// ---------- host launcher ----------
extern "C" void kernel_launch(void* const* d_in, const int* in_sizes, int n_in,
                              void* d_out, int out_size, void* d_ws, size_t ws_size,
                              hipStream_t stream) {
  (void)in_sizes; (void)n_in; (void)out_size; (void)ws_size;
  const float* x_t     = (const float*)d_in[0];
  const float* h_prev  = (const float*)d_in[1];
  const float* W_ih    = (const float*)d_in[2];
  const float* W_hh    = (const float*)d_in[3];
  const float* b_ih    = (const float*)d_in[4];
  const float* b_hh    = (const float*)d_in[5];
  const float* ln_g    = (const float*)d_in[6];
  const float* ln_b    = (const float*)d_in[7];
  const float* Wq_proj = (const float*)d_in[8];
  const float* bq_proj = (const float*)d_in[9];
  const float* W_q     = (const float*)d_in[10];
  const float* W_k     = (const float*)d_in[11];
  const float* W_v     = (const float*)d_in[12];
  const float* b_q     = (const float*)d_in[13];
  const float* b_k     = (const float*)d_in[14];
  const float* b_v     = (const float*)d_in[15];
  const float* W_o     = (const float*)d_in[16];
  const float* b_o     = (const float*)d_in[17];
  const float* aln_g   = (const float*)d_in[18];
  const float* aln_b   = (const float*)d_in[19];
  const float* W_gate  = (const float*)d_in[20];
  const float* b_gate  = (const float*)d_in[21];
  const float* W_read  = (const float*)d_in[22];
  const float* b_read  = (const float*)d_in[23];

  float* out        = (float*)d_out;
  float* h_new_out  = out;                               // (B,K,H)
  float* y_hat_out  = out + (i64)cB*cK*cH;               // (B,O)
  float* gt_out     = y_hat_out + (i64)cB*cO;            // (B,K)
  float* logits_out = gt_out + (i64)cB*cK;               // (B,K)

  char* wp = (char*)d_ws;
  auto alloc = [&](i64 bytes) -> char* {
    char* p = wp; wp += (bytes + 255) & ~(i64)255; return p;
  };
  bf16*  x_bf     = (bf16*) alloc((i64)cB*cD*2);
  bf16*  hp_bf    = (bf16*) alloc((i64)cB*cK*cH*2);
  bf16*  wih_bf   = (bf16*) alloc((i64)cK*c3H*cD*2);
  bf16*  whh_bf   = (bf16*) alloc((i64)cK*c3H*cH*2);
  bf16*  wqp_bf   = (bf16*) alloc((i64)cH*cD*2);
  bf16*  wq_bf    = (bf16*) alloc((i64)cH*cH*2);
  bf16*  wk_bf    = (bf16*) alloc((i64)cH*cH*2);
  bf16*  wv_bf    = (bf16*) alloc((i64)cH*cH*2);
  bf16*  wo_bf    = (bf16*) alloc((i64)cH*cH*2);
  bf16*  wr_bf    = (bf16*) alloc((i64)cO*cH*2);
  bf16*  hn_bf    = (bf16*) alloc((i64)cB*cK*cH*2);
  float* query_f  = (float*)alloc((i64)cB*cH*4);
  bf16*  query_bf = (bf16*) alloc((i64)cB*cH*2);
  float* q_f      = (float*)alloc((i64)cB*cH*4);
  bf16*  k_bf     = (bf16*) alloc((i64)cB*cK*cH*2);
  bf16*  v_bf     = (bf16*) alloc((i64)cB*cK*cH*2);
  bf16*  ctx_bf   = (bf16*) alloc((i64)cB*cH*2);
  float* attn_f   = (float*)alloc((i64)cB*cH*4);
  bf16*  mix_bf   = (bf16*) alloc((i64)cB*cH*2);

  auto cast = [&](const float* src, bf16* dst, i64 n) {
    cast_f32_bf16_kernel<<<dim3((unsigned)((n + 255) / 256)), dim3(256), 0, stream>>>(src, dst, n);
  };
  cast(x_t,     x_bf,   (i64)cB*cD);
  cast(h_prev,  hp_bf,  (i64)cB*cK*cH);
  cast(W_ih,    wih_bf, (i64)cK*c3H*cD);
  cast(W_hh,    whh_bf, (i64)cK*c3H*cH);
  cast(Wq_proj, wqp_bf, (i64)cH*cD);
  cast(W_q,     wq_bf,  (i64)cH*cH);
  cast(W_k,     wk_bf,  (i64)cH*cH);
  cast(W_v,     wv_bf,  (i64)cH*cH);
  cast(W_o,     wo_bf,  (i64)cH*cH);
  cast(W_read,  wr_bf,  (i64)cO*cH);

  // 1. fused GRU gates -> pre-LN h_new in d_out
  gru_gates_wmma_kernel<<<dim3(cB/32, cH/16, cK), dim3(32), 0, stream>>>(
      x_bf, hp_bf, wih_bf, whh_bf, b_ih, b_hh, h_prev, h_new_out);

  // 2. per-expert LayerNorm in place + bf16 copy
  expert_ln_kernel<<<dim3(cB*cK), dim3(256), 0, stream>>>(h_new_out, ln_g, ln_b, hn_bf);

  // 3. query = x @ Wq_proj^T + bq   (f32 + bf16)
  gemm_wmma_kernel<<<dim3(cB/32, cH/64), dim3(32), 0, stream>>>(
      x_bf, (i64)cD, wqp_bf, (i64)cD, bq_proj, query_f, query_bf, (i64)cH, cD);
  // 4. q = query @ W_q^T + b_q      (f32)
  gemm_wmma_kernel<<<dim3(cB/32, cH/64), dim3(32), 0, stream>>>(
      query_bf, (i64)cH, wq_bf, (i64)cH, b_q, q_f, (bf16*)nullptr, (i64)cH, cH);
  // 5. k = h_new @ W_k^T + b_k over all (b,k) rows (bf16)
  gemm_wmma_kernel<<<dim3(cB*cK/32, cH/64), dim3(32), 0, stream>>>(
      hn_bf, (i64)cH, wk_bf, (i64)cH, b_k, (float*)nullptr, k_bf, (i64)cH, cH);
  // 6. v = h_new @ W_v^T + b_v (bf16)
  gemm_wmma_kernel<<<dim3(cB*cK/32, cH/64), dim3(32), 0, stream>>>(
      hn_bf, (i64)cH, wv_bf, (i64)cH, b_v, (float*)nullptr, v_bf, (i64)cH, cH);

  // 7. attention over K experts -> ctx (bf16)
  attn_kernel<<<dim3(cB), dim3(256), 0, stream>>>(q_f, k_bf, v_bf, ctx_bf);

  // 8. attn_out = ctx @ W_o^T + b_o (f32)
  gemm_wmma_kernel<<<dim3(cB/32, cH/64), dim3(32), 0, stream>>>(
      ctx_bf, (i64)cH, wo_bf, (i64)cH, b_o, attn_f, (bf16*)nullptr, (i64)cH, cH);

  // 9. residual+LN, gate logits + softmax, expert mix
  gate_mix_kernel<<<dim3(cB), dim3(256), 0, stream>>>(
      query_f, attn_f, aln_g, aln_b, W_gate, b_gate, h_new_out,
      gt_out, logits_out, mix_bf);

  // 10. y_hat = mixed_h @ W_read^T + b_read (f32, straight into d_out)
  gemm_wmma_kernel<<<dim3(cB/32, cO/64), dim3(32), 0, stream>>>(
      mix_bf, (i64)cH, wr_bf, (i64)cH, b_read, y_hat_out, (bf16*)nullptr, (i64)cO, cH);
}
